// AttentionBlock_43808666419569
// MI455X (gfx1250) — matrix operand order repaired and checked by reading, not compiled, not run on hardware
//
#include <hip/hip_runtime.h>

#define CH 256
#define NTOK 4096
#define GROUPS 8
#define CPG 32
#define BATCH 4
#define KT 64            // flash-attention key tile

typedef _Float16 h16;
typedef __attribute__((ext_vector_type(16))) _Float16 v16h;
typedef __attribute__((ext_vector_type(8)))  _Float16 v8h;
typedef __attribute__((ext_vector_type(8)))  float    v8f;
typedef __attribute__((ext_vector_type(4)))  unsigned int u32x4;
typedef __attribute__((ext_vector_type(8)))  int i32x8;
typedef __attribute__((ext_vector_type(4)))  int i32x4;

__device__ __forceinline__ v8f wmma16(v16h a, v16h b, v8f c) {
  return __builtin_amdgcn_wmma_f32_16x16x32_f16(false, a, false, b, (short)0, c, false, false);
}

// A fragment (16M x 32K) from [M][ld] storage, M-major. lane: M = lane&15, hs = lane>>4.
// element e: K = 8*hs + e (e<8), K = 16 + 8*hs + (e-8) (e>=8)  -> two contiguous v8h loads.
__device__ __forceinline__ v16h load_a_frag(const h16* base, int ld, int row0,
                                            int k0, int lane) {
  int m = lane & 15, hs = lane >> 4;
  const h16* p = base + (size_t)(row0 + m) * ld + k0 + 8 * hs;
  v8h lo = *(const v8h*)p;
  v8h hi = *(const v8h*)(p + 16);
  return __builtin_shufflevector(lo, hi, 0,1,2,3,4,5,6,7,8,9,10,11,12,13,14,15);
}

// B fragment (32K x 16N) from N-major storage [N][ld]. lane: N = lane&15, hs = lane>>4.
// element e: K = 8*hs + e (e<8), K = 16 + 8*hs + (e-8) (e>=8)  -> two contiguous v8h loads.
__device__ __forceinline__ v16h load_b_frag(const h16* base, int ld, int col0,
                                            int k0, int lane) {
  int n = lane & 15, hs = lane >> 4;
  const h16* p = base + (size_t)(col0 + n) * ld + k0 + 8 * hs;
  v8h lo = *(const v8h*)p;
  v8h hi = *(const v8h*)(p + 16);
  return __builtin_shufflevector(lo, hi, 0,1,2,3,4,5,6,7,8,9,10,11,12,13,14,15);
}

// ---------------------------------------------------------------------------
// TDM: one-instruction 2D tile DMA global->LDS (CDNA5 Tensor Data Mover).
// Descriptor packed per ISA 8.3/8.4: count=1, type=2, data_size=2B,
// tile_d0 = elems per row (contig), tile_d1 = rows, stride0 = row stride.
// Groups 2/3 zero (2-D tile). Tracked by TENSORcnt.
// This toolchain's builtin is the 6-arg form (clang-23 / therock-10.0).
// ---------------------------------------------------------------------------
__device__ __forceinline__ void tdm_load_2d(unsigned lds_addr, const h16* gptr,
                                            unsigned tensor_d0, unsigned tensor_d1,
                                            unsigned stride0,
                                            unsigned tile_d0, unsigned tile_d1) {
  unsigned long long ga = (unsigned long long)(uintptr_t)gptr;
  u32x4 g0;
  g0[0] = 1u;                                        // count=1, user mode
  g0[1] = lds_addr;                                  // LDS byte address
  g0[2] = (unsigned)(ga & 0xFFFFFFFFu);              // global_addr[31:0]
  g0[3] = (unsigned)((ga >> 32) & 0x01FFFFFFu) | 0x80000000u;  // ga[56:32] | type=2
  i32x8 g1;
  g1[0] = (int)(1u << 16);                           // data_size=1 -> 2 bytes
  g1[1] = (int)((tensor_d0 & 0xFFFFu) << 16);        // tensor_dim0[15:0] @ bit48
  g1[2] = (int)(((tensor_d0 >> 16) & 0xFFFFu) | ((tensor_d1 & 0xFFFFu) << 16));
  g1[3] = (int)(((tensor_d1 >> 16) & 0xFFFFu) | (tile_d0 << 16));
  g1[4] = (int)(tile_d1 & 0xFFFFu);                  // tile_dim1 | tile_dim2=0
  g1[5] = (int)stride0;                              // tensor_dim0_stride lo32
  g1[6] = 0;
  g1[7] = 0;
  i32x4 gz4 = {0, 0, 0, 0};
  i32x8 gz8 = {0, 0, 0, 0, 0, 0, 0, 0};
  __builtin_amdgcn_tensor_load_to_lds(g0, g1, gz4, gz4, gz8, 0);
}

// ---------------------------------------------------------------------------
// 1) Weight convert to f16 (softmax scale 1/16 folded into wq/bq), biases to ws
// ---------------------------------------------------------------------------
__global__ __launch_bounds__(256) void convert_weights(
    const float* __restrict__ wq, const float* __restrict__ wk,
    const float* __restrict__ wv, const float* __restrict__ wp,
    const float* __restrict__ bq, const float* __restrict__ bk,
    const float* __restrict__ bv, const float* __restrict__ bp,
    h16* __restrict__ wqh, h16* __restrict__ wkh,
    h16* __restrict__ wvh, h16* __restrict__ wph,
    float* __restrict__ biases) {
  const float SC = 0.0625f;  // 1/sqrt(256)
  int i = blockIdx.x * 256 + threadIdx.x;
  if (i < CH * CH) {
    wqh[i] = (h16)(wq[i] * SC);
    wkh[i] = (h16)wk[i];
    wvh[i] = (h16)wv[i];
    wph[i] = (h16)wp[i];
  }
  if (i < CH) {
    biases[i]          = bq[i] * SC;
    biases[CH + i]     = bk[i];
    biases[2 * CH + i] = bv[i];
    biases[3 * CH + i] = bp[i];
  }
}

// ---------------------------------------------------------------------------
// 2) GroupNorm: x[b][c][n] f32 -> xn_t[b][n][c] f16 (token-major), LDS bounce
// ---------------------------------------------------------------------------
__global__ __launch_bounds__(256) void groupnorm_kernel(
    const float* __restrict__ x, const float* __restrict__ gamma,
    const float* __restrict__ beta, h16* __restrict__ Xt) {
  __shared__ float redA[256];
  __shared__ float redB[256];
  __shared__ h16 tile[64 * CPG];  // 64 tokens x 32 channels
  int b = blockIdx.x >> 3, g = blockIdx.x & 7;
  int tid = threadIdx.x;
  const float* xg = x + ((size_t)(b * CH) + g * CPG) * NTOK;

  float s = 0.f, ss = 0.f;
  for (int i = tid; i < CPG * NTOK; i += 256) {
    float v = xg[i];
    s += v; ss += v * v;
  }
  redA[tid] = s; redB[tid] = ss;
  __syncthreads();
  for (int st = 128; st > 0; st >>= 1) {
    if (tid < st) { redA[tid] += redA[tid + st]; redB[tid] += redB[tid + st]; }
    __syncthreads();
  }
  const float inv_cnt = 1.f / (float)(CPG * NTOK);
  float mean = redA[0] * inv_cnt;
  float var = redB[0] * inv_cnt - mean * mean;
  float rstd = rsqrtf(var + 1e-5f);
  __syncthreads();

  for (int n0 = 0; n0 < NTOK; n0 += 64) {
#pragma unroll
    for (int kk = 0; kk < 8; ++kk) {        // coalesced read 32c x 64n
      int idx = kk * 256 + tid;
      int c = idx >> 6, nl = idx & 63;
      float v = xg[(size_t)c * NTOK + n0 + nl];
      float xn = (v - mean) * rstd * gamma[g * CPG + c] + beta[g * CPG + c];
      tile[nl * CPG + c] = (h16)xn;
    }
    __syncthreads();
#pragma unroll
    for (int kk = 0; kk < 8; ++kk) {        // coalesced token-major write
      int idx = kk * 256 + tid;
      int nl = idx >> 5, c = idx & 31;
      Xt[((size_t)b * NTOK + n0 + nl) * CH + g * CPG + c] = tile[nl * CPG + c];
    }
    __syncthreads();
  }
}

// ---------------------------------------------------------------------------
// 3) Q/K GEMM: Yt[b][n][o] = sum_c Xt[b][n][c] * W[o][c] + bias[o]   (f16 out)
// ---------------------------------------------------------------------------
__global__ __launch_bounds__(256) void gemm_xw_tok(
    const h16* __restrict__ Xt, const h16* __restrict__ W,
    const float* __restrict__ bias, h16* __restrict__ Yt) {
  int b = blockIdx.z;
  int n0 = blockIdx.x * 128;
  int o0 = blockIdx.y * 128;
  int wave = threadIdx.x >> 5, lane = threadIdx.x & 31;
  int wm = wave & 3, wn = wave >> 2;
  const h16* Xb = Xt + (size_t)b * NTOK * CH;
  v8f acc[2][4] = {};
  for (int k0 = 0; k0 < CH; k0 += 32) {
    v16h a[2], bf[4];
#pragma unroll
    for (int i = 0; i < 2; ++i)
      a[i] = load_a_frag(Xb, CH, n0 + wm * 32 + i * 16, k0, lane);
#pragma unroll
    for (int j = 0; j < 4; ++j)
      bf[j] = load_b_frag(W, CH, o0 + wn * 64 + j * 16, k0, lane);
#pragma unroll
    for (int i = 0; i < 2; ++i)
#pragma unroll
      for (int j = 0; j < 4; ++j)
        acc[i][j] = wmma16(a[i], bf[j], acc[i][j]);
  }
  int lm = lane & 15, hs = lane >> 4;
#pragma unroll
  for (int i = 0; i < 2; ++i)
#pragma unroll
    for (int j = 0; j < 4; ++j) {
      int oc = o0 + wn * 64 + j * 16 + lm;
      float bvv = bias[oc];
#pragma unroll
      for (int r = 0; r < 8; ++r) {
        int nr = n0 + wm * 32 + i * 16 + r + 8 * hs;
        Yt[((size_t)b * NTOK + nr) * CH + oc] = (h16)(acc[i][j][r] + bvv);
      }
    }
}

// ---------------------------------------------------------------------------
// 4) V GEMM (channel-major out): Ycm[b][o][m] = sum_c W[o][c]*Xt[b][m][c]+bias
// ---------------------------------------------------------------------------
__global__ __launch_bounds__(256) void gemm_v_cm(
    const h16* __restrict__ W, const h16* __restrict__ Xt,
    const float* __restrict__ bias, h16* __restrict__ Ycm) {
  int b = blockIdx.z;
  int m0 = blockIdx.x * 128;
  int o0 = blockIdx.y * 128;
  int wave = threadIdx.x >> 5, lane = threadIdx.x & 31;
  int wo = wave & 3, wt = wave >> 2;
  const h16* Xb = Xt + (size_t)b * NTOK * CH;
  v8f acc[2][4] = {};
  for (int k0 = 0; k0 < CH; k0 += 32) {
    v16h a[2], bf[4];
#pragma unroll
    for (int i = 0; i < 2; ++i)
      a[i] = load_a_frag(W, CH, o0 + wo * 32 + i * 16, k0, lane);
#pragma unroll
    for (int j = 0; j < 4; ++j)
      bf[j] = load_b_frag(Xb, CH, m0 + wt * 64 + j * 16, k0, lane);
#pragma unroll
    for (int i = 0; i < 2; ++i)
#pragma unroll
      for (int j = 0; j < 4; ++j)
        acc[i][j] = wmma16(a[i], bf[j], acc[i][j]);
  }
  int lm = lane & 15, hs = lane >> 4;
#pragma unroll
  for (int i = 0; i < 2; ++i)
#pragma unroll
    for (int j = 0; j < 4; ++j) {
      int mc = m0 + wt * 64 + j * 16 + lm;
#pragma unroll
      for (int r = 0; r < 8; ++r) {
        int orow = o0 + wo * 32 + i * 16 + r + 8 * hs;
        Ycm[((size_t)b * CH + orow) * NTOK + mc] = (h16)(acc[i][j][r] + bias[orow]);
      }
    }
}

// ---------------------------------------------------------------------------
// 5) Flash attention with TDM double-buffered K/V staging.
//    LDS: K[2][64x256] (2x32KB) | V[2][256x64] (2x32KB) | P[8][16x64] (16KB).
//    Wave 0 drives the Tensor Data Mover; s_wait_tensorcnt + barrier publish.
//    Per 64-key tile, per wave: 32 WMMA (S=Q*K^T) + 32 WMMA (P*V^T).
// ---------------------------------------------------------------------------
#define K_OFF(i)  ((unsigned)(i) * 32768u)
#define V_OFF(i)  (65536u + (unsigned)(i) * 32768u)
#define P_OFF     131072u
#define SMEM_BYTES (131072u + 16384u)

__global__ __launch_bounds__(256) void flash_attn(
    const h16* __restrict__ Qt, const h16* __restrict__ Kt,
    const h16* __restrict__ Vcm, h16* __restrict__ Ot) {
  extern __shared__ char smem[];
  int b = blockIdx.y;
  int wave = threadIdx.x >> 5, lane = threadIdx.x & 31;
  int q0 = blockIdx.x * 128 + wave * 16;
  int lm = lane & 15, hs = lane >> 4;
  const h16* Qb = Qt + (size_t)b * NTOK * CH;
  const h16* Kb = Kt + (size_t)b * NTOK * CH;
  const h16* Vb = Vcm + (size_t)b * CH * NTOK;
  const unsigned lds0 = (unsigned)(uintptr_t)smem;   // LDS byte base for TDM
  h16* pb = (h16*)(smem + P_OFF) + wave * (16 * KT);

  v8f acc_o[16] = {};          // 16 queries x 256 channels, f32
  float mrow[8], lrow[8];
#pragma unroll
  for (int r = 0; r < 8; ++r) { mrow[r] = -1e30f; lrow[r] = 0.f; }

  v16h qf[8];                  // hoist all Q fragments (K=256 = 8 x 32)
#pragma unroll
  for (int kk = 0; kk < 8; ++kk) qf[kk] = load_a_frag(Qb, CH, q0, kk * 32, lane);

  const int NIT = NTOK / KT;
  if (wave == 0) {             // prologue: DMA first K/V tile
    tdm_load_2d(lds0 + K_OFF(0), Kb, CH, NTOK, CH, CH, KT);
    tdm_load_2d(lds0 + V_OFF(0), Vb, NTOK, CH, NTOK, KT, CH);
  }

  for (int it = 0; it < NIT; ++it) {
    int cur = it & 1;
    if (wave == 0) __builtin_amdgcn_s_wait_tensorcnt(0);  // tile `it` landed
    __syncthreads();                                      // publish to block
    if (wave == 0 && it + 1 < NIT) {                      // prefetch tile it+1
      int m1 = (it + 1) * KT, nxt = cur ^ 1;
      tdm_load_2d(lds0 + K_OFF(nxt), Kb + (size_t)m1 * CH, CH, NTOK, CH, CH, KT);
      tdm_load_2d(lds0 + V_OFF(nxt), Vb + m1, NTOK, CH, NTOK, KT, CH);
    }
    const h16* ktile = (const h16*)(smem + K_OFF(cur));   // [KT][CH]
    const h16* vtile = (const h16*)(smem + V_OFF(cur));   // [CH][KT]

    // S = Q * K^T  -> [16 x 64]
    v8f s[4] = {};
#pragma unroll
    for (int kk = 0; kk < 8; ++kk)
#pragma unroll
      for (int mj = 0; mj < 4; ++mj) {
        v16h bk = load_b_frag(ktile, CH, mj * 16, kk * 32, lane);
        s[mj] = wmma16(qf[kk], bk, s[mj]);
      }

    // online softmax (masks 1/2/4/8 stay inside each 16-lane half)
    float pscale[8];
#pragma unroll
    for (int r = 0; r < 8; ++r) {
      float v = fmaxf(fmaxf(s[0][r], s[1][r]), fmaxf(s[2][r], s[3][r]));
      v = fmaxf(v, __shfl_xor(v, 1));
      v = fmaxf(v, __shfl_xor(v, 2));
      v = fmaxf(v, __shfl_xor(v, 4));
      v = fmaxf(v, __shfl_xor(v, 8));
      float mnew = fmaxf(mrow[r], v);
      float corr = __expf(mrow[r] - mnew);
      float rs = 0.f;
#pragma unroll
      for (int mj = 0; mj < 4; ++mj) {
        float p = __expf(s[mj][r] - mnew);
        s[mj][r] = p;
        rs += p;
      }
      rs += __shfl_xor(rs, 1);
      rs += __shfl_xor(rs, 2);
      rs += __shfl_xor(rs, 4);
      rs += __shfl_xor(rs, 8);
      mrow[r] = mnew;
      lrow[r] = lrow[r] * corr + rs;
      pscale[r] = corr;
    }
#pragma unroll
    for (int ct = 0; ct < 16; ++ct)
#pragma unroll
      for (int r = 0; r < 8; ++r) acc_o[ct][r] *= pscale[r];

    // P: D-layout -> LDS [q][KT] -> A-layout fragments (same-wave bounce)
#pragma unroll
    for (int mj = 0; mj < 4; ++mj)
#pragma unroll
      for (int r = 0; r < 8; ++r)
        pb[(r + 8 * hs) * KT + mj * 16 + lm] = (h16)s[mj][r];
    asm volatile("s_wait_dscnt 0" ::: "memory");

    // O += P * V^T
#pragma unroll
    for (int kp = 0; kp < KT; kp += 32) {
      v16h ap;
      {
        const h16* p = pb + lm * KT + kp + 8 * hs;
        v8h lo = *(const v8h*)p;
        v8h hi = *(const v8h*)(p + 16);
        ap = __builtin_shufflevector(lo, hi, 0,1,2,3,4,5,6,7,8,9,10,11,12,13,14,15);
      }
#pragma unroll
      for (int ct = 0; ct < 16; ++ct) {
        v16h bv = load_b_frag(vtile, KT, ct * 16, kp, lane);
        acc_o[ct] = wmma16(ap, bv, acc_o[ct]);
      }
    }
    __syncthreads();   // block done with tile `it` before it is overwritten
  }

  float rinv[8];
#pragma unroll
  for (int r = 0; r < 8; ++r) rinv[r] = 1.f / lrow[r];
#pragma unroll
  for (int ct = 0; ct < 16; ++ct)
#pragma unroll
    for (int r = 0; r < 8; ++r) {
      int q = q0 + r + 8 * hs;
      Ot[((size_t)b * NTOK + q) * CH + ct * 16 + lm] = (h16)(acc_o[ct][r] * rinv[r]);
    }
}

// ---------------------------------------------------------------------------
// 6) Projection + bias + residual: out[b][o][n] f32
// ---------------------------------------------------------------------------
__global__ __launch_bounds__(256) void gemm_proj(
    const h16* __restrict__ Wph, const h16* __restrict__ Ot,
    const float* __restrict__ bias, const float* __restrict__ xres,
    float* __restrict__ out) {
  int b = blockIdx.z;
  int n0 = blockIdx.x * 128;
  int o0 = blockIdx.y * 128;
  int wave = threadIdx.x >> 5, lane = threadIdx.x & 31;
  int wo = wave & 3, wt = wave >> 2;
  const h16* Ob = Ot + (size_t)b * NTOK * CH;
  v8f acc[2][4] = {};
  for (int k0 = 0; k0 < CH; k0 += 32) {
    v16h a[2], bf[4];
#pragma unroll
    for (int i = 0; i < 2; ++i)
      a[i] = load_a_frag(Wph, CH, o0 + wo * 32 + i * 16, k0, lane);
#pragma unroll
    for (int j = 0; j < 4; ++j)
      bf[j] = load_b_frag(Ob, CH, n0 + wt * 64 + j * 16, k0, lane);
#pragma unroll
    for (int i = 0; i < 2; ++i)
#pragma unroll
      for (int j = 0; j < 4; ++j)
        acc[i][j] = wmma16(a[i], bf[j], acc[i][j]);
  }
  int lm = lane & 15, hs = lane >> 4;
#pragma unroll
  for (int i = 0; i < 2; ++i)
#pragma unroll
    for (int j = 0; j < 4; ++j) {
      int nc = n0 + wt * 64 + j * 16 + lm;
#pragma unroll
      for (int r = 0; r < 8; ++r) {
        int orow = o0 + wo * 32 + i * 16 + r + 8 * hs;
        size_t idx = ((size_t)b * CH + orow) * NTOK + nc;
        out[idx] = acc[i][j][r] + bias[orow] + xres[idx];
      }
    }
}

// ---------------------------------------------------------------------------
extern "C" void kernel_launch(void* const* d_in, const int* in_sizes, int n_in,
                              void* d_out, int out_size, void* d_ws, size_t ws_size,
                              hipStream_t stream) {
  const float* x     = (const float*)d_in[0];
  const float* gamma = (const float*)d_in[1];
  const float* beta  = (const float*)d_in[2];
  const float* wq = (const float*)d_in[3];  const float* bq = (const float*)d_in[4];
  const float* wk = (const float*)d_in[5];  const float* bk = (const float*)d_in[6];
  const float* wv = (const float*)d_in[7];  const float* bv = (const float*)d_in[8];
  const float* wp = (const float*)d_in[9];  const float* bp = (const float*)d_in[10];
  float* out = (float*)d_out;

  char* ws = (char*)d_ws;
  const size_t WSZ = (size_t)CH * CH * sizeof(h16);        // 128 KB each
  h16* wqh = (h16*)(ws);
  h16* wkh = (h16*)(ws + WSZ);
  h16* wvh = (h16*)(ws + 2 * WSZ);
  h16* wph = (h16*)(ws + 3 * WSZ);
  float* biases = (float*)(ws + 4 * WSZ);                  // 4 x 256 f32
  const size_t ACT = (size_t)BATCH * NTOK * CH * sizeof(h16);  // 8 MB each
  const size_t MB = 1u << 20;
  h16* xn_t = (h16*)(ws + 1 * MB);
  h16* q_t  = (h16*)(ws + 1 * MB + ACT);
  h16* k_t  = (h16*)(ws + 1 * MB + 2 * ACT);
  h16* v_cm = (h16*)(ws + 1 * MB + 3 * ACT);
  h16* o_t  = (h16*)(ws + 1 * MB + 4 * ACT);

  convert_weights<<<dim3(CH * CH / 256), 256, 0, stream>>>(
      wq, wk, wv, wp, bq, bk, bv, bp, wqh, wkh, wvh, wph, biases);

  groupnorm_kernel<<<dim3(BATCH * GROUPS), 256, 0, stream>>>(x, gamma, beta, xn_t);

  dim3 ggrid(NTOK / 128, CH / 128, BATCH);
  gemm_xw_tok<<<ggrid, 256, 0, stream>>>(xn_t, wqh, biases, q_t);
  gemm_xw_tok<<<ggrid, 256, 0, stream>>>(xn_t, wkh, biases + CH, k_t);
  gemm_v_cm <<<ggrid, 256, 0, stream>>>(wvh, xn_t, biases + 2 * CH, v_cm);

  flash_attn<<<dim3(NTOK / 128, BATCH), 256, SMEM_BYTES, stream>>>(q_t, k_t, v_cm, o_t);

  gemm_proj<<<ggrid, 256, 0, stream>>>(wph, o_t, biases + 3 * CH, x, out);
}
